// Bidirectional_RNN_3728031613356
// MI455X (gfx1250) — matrix operand order repaired and checked by reading, not compiled, not run on hardware
//
#include <hip/hip_runtime.h>

// Problem constants (match reference)
#define BB   8
#define TT   4096
#define DD   256
#define HH   256
#define H3   768       // 3*H
#define H2   512       // 2*H
#define BT   32768     // B*T

typedef __attribute__((ext_vector_type(16))) __bf16 v16bf;
typedef __attribute__((ext_vector_type(8)))  __bf16 v8bf;
typedef __attribute__((ext_vector_type(8)))  float  v8f;

static __device__ __forceinline__ v8f zero8() {
  v8f z = {0.f, 0.f, 0.f, 0.f, 0.f, 0.f, 0.f, 0.f};
  return z;
}

// A-fragment (16x32 bf16, row-major source, row pointer given):
// lanes 0-15 (lane_hi=0): K = k0+0..7 (v0..3), k0+16..23 (v4..7)
// lanes 16-31 (lane_hi=1): K = k0+8..15,       k0+24..31
static __device__ __forceinline__ v16bf load_a_frag(const __bf16* row, int lane_hi, int k0) {
  const v8bf lo = *(const v8bf*)(row + k0 + lane_hi * 8);
  const v8bf hi = *(const v8bf*)(row + k0 + 16 + lane_hi * 8);
  return __builtin_shufflevector(lo, hi, 0,1,2,3,4,5,6,7,8,9,10,11,12,13,14,15);
}

// B-fragment (32x16 bf16) loaded from transposed weights WT[n][k] (row = column n):
// lanes 0-15: K = k0+0..7, k0+16..23 ; lanes 16-31: K = k0+8..15, k0+24..31 (2 per VGPR)
static __device__ __forceinline__ v16bf load_b_frag(const __bf16* row, int lane_hi, int k0) {
  const int kb = k0 + lane_hi * 8;
  const v8bf lo = *(const v8bf*)(row + kb);
  const v8bf hi = *(const v8bf*)(row + kb + 16);
  return __builtin_shufflevector(lo, hi, 0,1,2,3,4,5,6,7,8,9,10,11,12,13,14,15);
}

static __device__ __forceinline__ v8f wmma_bf16(v16bf a, v16bf b, v8f c) {
  return __builtin_amdgcn_wmma_f32_16x16x32_bf16(false, a, false, b, (short)0, c, false, false);
}

static __device__ __forceinline__ float sigmoidf_fast(float x) {
  return 1.f / (1.f + __expf(-x));
}

// ---------------------------------------------------------------- prep kernels

__global__ void k_cvt_bf16(__bf16* __restrict__ dst, const float* __restrict__ src, int n) {
  int i = blockIdx.x * blockDim.x + threadIdx.x;
  if (i < n) dst[i] = (__bf16)src[i];
}

// src row-major [K][N] fp32 -> dst row-major [N][K] bf16
__global__ void k_transpose_bf16(__bf16* __restrict__ dst, const float* __restrict__ src,
                                 int K, int N) {
  int i = blockIdx.x * blockDim.x + threadIdx.x;
  if (i < K * N) {
    int k = i / N, n = i % N;
    dst[(size_t)n * K + k] = (__bf16)src[i];
  }
}

// ------------------------------------------------------- input-projection GEMM
// XG[dir] = Xbf[BT,D] @ Wi[dir] + bi[dir]   (output fp32 [BT, 3H])
__global__ __launch_bounds__(256) void k_gemm_in(
    const __bf16* __restrict__ Xbf,
    const __bf16* __restrict__ WiT_f, const __bf16* __restrict__ WiT_b,  // [H3][D]
    const float* __restrict__ bi_f, const float* __restrict__ bi_b,
    float* __restrict__ XG_f, float* __restrict__ XG_b) {
  const int dir = blockIdx.y;
  const __bf16* WT = dir ? WiT_b : WiT_f;
  const float*  bi = dir ? bi_b : bi_f;
  float*        XG = dir ? XG_b : XG_f;

  const int lane = threadIdx.x & 31;
  const int wave = threadIdx.x >> 5;
  const int tile = blockIdx.x * 8 + wave;        // 2048*48 tiles
  const int nt = tile % 48, mt = tile / 48;
  const int lane_lo = lane & 15, lane_hi = lane >> 4;

  const __bf16* arow = Xbf + (size_t)(mt * 16 + lane_lo) * DD;
  const __bf16* brow = WT + (size_t)(nt * 16 + lane_lo) * DD;

  v8f acc = zero8();
#pragma unroll
  for (int kk = 0; kk < DD / 32; ++kk) {
    v16bf a = load_a_frag(arow, lane_hi, kk * 32);
    v16bf b = load_b_frag(brow, lane_hi, kk * 32);
    acc = wmma_bf16(a, b, acc);
  }

  const int n = nt * 16 + lane_lo;
  const float bias = bi[n];
  float* out = XG + (size_t)(mt * 16) * H3 + n;
#pragma unroll
  for (int r = 0; r < 8; ++r) {
    const int m = r + lane_hi * 8;
    out[(size_t)m * H3] = acc[r] + bias;
  }
}

// --------------------------------------------------------- persistent GRU scan
// One workgroup per direction. 16 waves; wave w owns N-tiles {3w,3w+1,3w+2}
// of Wh^T, register-resident bf16 fragments for all 4096 steps.
__global__ __launch_bounds__(512) void k_scan(
    const float* __restrict__ XG_f, const float* __restrict__ XG_b,
    const __bf16* __restrict__ WhT_f, const __bf16* __restrict__ WhT_b,  // [H3][H]
    const float* __restrict__ bhn_f, const float* __restrict__ bhn_b,
    __bf16* __restrict__ outc) {                                         // [BT, 2H]
  const int dir = blockIdx.x;
  const float*  XG  = dir ? XG_b  : XG_f;
  const __bf16* WhT = dir ? WhT_b : WhT_f;
  const float*  bhn = dir ? bhn_b : bhn_f;

  __shared__ __bf16 hbf[16][HH + 8];   // padded rows; rows 8..15 stay zero
  __shared__ float  h32[8][HH];
  __shared__ float  gh[8][H3];
  __shared__ float  sbhn[HH];

  const int tid = threadIdx.x;
  const int lane = tid & 31;
  const int wave = tid >> 5;
  const int lane_lo = lane & 15, lane_hi = lane >> 4;

  for (int i = tid; i < 16 * (HH + 8); i += 512) ((__bf16*)hbf)[i] = (__bf16)0.f;
  for (int i = tid; i < 8 * HH; i += 512) ((float*)h32)[i] = 0.f;
  for (int i = tid; i < HH; i += 512) sbhn[i] = bhn[i];
  __syncthreads();

  // register-resident hidden weights: 3 tiles x 8 k-frags per wave
  v16bf Breg[3][8];
#pragma unroll
  for (int j = 0; j < 3; ++j) {
    const int n = (3 * wave + j) * 16 + lane_lo;
    const __bf16* brow = WhT + (size_t)n * HH;
#pragma unroll
    for (int kk = 0; kk < 8; ++kk) Breg[j][kk] = load_b_frag(brow, lane_hi, kk * 32);
  }

  for (int step = 0; step < TT; ++step) {
    const int t = dir ? (TT - 1 - step) : step;

    // gh = h @ Wh  (rows 8..15 of A are zero)
    v8f acc[3];
    acc[0] = zero8(); acc[1] = zero8(); acc[2] = zero8();
#pragma unroll
    for (int kk = 0; kk < 8; ++kk) {
      v16bf a = load_a_frag(&hbf[lane_lo][0], lane_hi, kk * 32);
#pragma unroll
      for (int j = 0; j < 3; ++j) acc[j] = wmma_bf16(a, Breg[j][kk], acc[j]);
    }
    if (lane < 16) {
#pragma unroll
      for (int j = 0; j < 3; ++j) {
        const int n = (3 * wave + j) * 16 + lane_lo;
#pragma unroll
        for (int r = 0; r < 8; ++r) gh[r][n] = acc[j][r];  // rows 0..7 = batch
      }
    }
    __syncthreads();

    // gate math + hidden update (2048 units over 512 threads)
    for (int u = tid; u < 8 * HH; u += 512) {
      const int b = u >> 8, j = u & (HH - 1);
      const size_t ro = ((size_t)b * TT + t) * H3;
      const float gi_r = XG[ro + j];
      const float gi_z = XG[ro + HH + j];
      const float gi_n = XG[ro + 2 * HH + j];
      const float r = sigmoidf_fast(gi_r + gh[b][j]);
      const float z = sigmoidf_fast(gi_z + gh[b][HH + j]);
      const float nn = tanhf(gi_n + r * (gh[b][2 * HH + j] + sbhn[j]));
      const float hn = (1.f - z) * nn + z * h32[b][j];
      h32[b][j] = hn;
      hbf[b][j] = (__bf16)hn;
      outc[((size_t)b * TT + t) * H2 + dir * HH + j] = (__bf16)hn;
      const int t2 = dir ? (t - 1) : (t + 1);
      if ((unsigned)t2 < TT)  // warm cache for the next (serial) step
        __builtin_prefetch(&XG[((size_t)b * TT + t2) * H3 + j], 0, 0);
    }
    __syncthreads();
  }
}

// ----------------------------------------------------------------- heads GEMM
// [h | J] = outc[BT,2H] @ [Wm | Wv] (+bias, exp on J), written straight to d_out:
// d_out[0 .. BT*H)        = J_diag = exp(outc@Wv + bv)
// d_out[BT*H .. 2*BT*H)   = h      = outc@Wm + bm
__global__ __launch_bounds__(256) void k_head(
    const __bf16* __restrict__ outc,   // [BT, 2H]
    const __bf16* __restrict__ WmvT,   // [2H n][2H k]; rows 0..255 Wm, 256..511 Wv
    const float* __restrict__ bm, const float* __restrict__ bv,
    float* __restrict__ out) {
  const int lane = threadIdx.x & 31;
  const int wave = threadIdx.x >> 5;
  const int tile = blockIdx.x * 8 + wave;     // 2048*32 tiles
  const int nt = tile & 31, mt = tile >> 5;
  const int lane_lo = lane & 15, lane_hi = lane >> 4;

  const __bf16* arow = outc + (size_t)(mt * 16 + lane_lo) * H2;
  const __bf16* brow = WmvT + (size_t)(nt * 16 + lane_lo) * H2;

  v8f acc = zero8();
#pragma unroll
  for (int kk = 0; kk < H2 / 32; ++kk) {
    v16bf a = load_a_frag(arow, lane_hi, kk * 32);
    v16bf b = load_b_frag(brow, lane_hi, kk * 32);
    acc = wmma_bf16(a, b, acc);
  }

  const int n = nt * 16 + lane_lo;
  const bool isMean = (n < HH);
  const int ncol = isMean ? n : (n - HH);
  const float bias = isMean ? bm[ncol] : bv[ncol];
  const size_t base = isMean ? (size_t)BT * HH : 0;
#pragma unroll
  for (int r = 0; r < 8; ++r) {
    const int m = r + lane_hi * 8;
    const size_t row = (size_t)(mt * 16 + m);
    const float v = acc[r] + bias;
    out[base + row * HH + ncol] = isMean ? v : __expf(v);
  }
}

// ------------------------------------------------------------------- launcher
extern "C" void kernel_launch(void* const* d_in, const int* in_sizes, int n_in,
                              void* d_out, int out_size, void* d_ws, size_t ws_size,
                              hipStream_t stream) {
  const float* X    = (const float*)d_in[0];
  const float* Wi_f = (const float*)d_in[1];
  const float* bi_f = (const float*)d_in[2];
  const float* Wh_f = (const float*)d_in[3];
  const float* bhnf = (const float*)d_in[4];
  const float* Wi_b = (const float*)d_in[5];
  const float* bi_b = (const float*)d_in[6];
  const float* Wh_b = (const float*)d_in[7];
  const float* bhnb = (const float*)d_in[8];
  const float* Wm   = (const float*)d_in[9];
  const float* bm   = (const float*)d_in[10];
  const float* Wv   = (const float*)d_in[11];
  const float* bv   = (const float*)d_in[12];
  float* out = (float*)d_out;

  // workspace carve-up (256B aligned)
  char* p = (char*)d_ws;
  auto carve = [&](size_t bytes) {
    char* r = p;
    p += (bytes + 255) & ~(size_t)255;
    return r;
  };
  __bf16* XBF   = (__bf16*)carve((size_t)BT * DD * 2);
  __bf16* WIT_F = (__bf16*)carve((size_t)H3 * DD * 2);
  __bf16* WIT_B = (__bf16*)carve((size_t)H3 * DD * 2);
  __bf16* WHT_F = (__bf16*)carve((size_t)H3 * HH * 2);
  __bf16* WHT_B = (__bf16*)carve((size_t)H3 * HH * 2);
  __bf16* WMVT  = (__bf16*)carve((size_t)H2 * H2 * 2);
  float*  XG_F  = (float*)carve((size_t)BT * H3 * 4);
  float*  XG_B  = (float*)carve((size_t)BT * H3 * 4);
  __bf16* OUTC  = (__bf16*)carve((size_t)BT * H2 * 2);
  (void)ws_size; (void)in_sizes; (void)n_in; (void)out_size;

  // precision prep: bf16 copies / transposes
  k_cvt_bf16<<<(BT * DD + 255) / 256, 256, 0, stream>>>(XBF, X, BT * DD);
  k_transpose_bf16<<<(DD * H3 + 255) / 256, 256, 0, stream>>>(WIT_F, Wi_f, DD, H3);
  k_transpose_bf16<<<(DD * H3 + 255) / 256, 256, 0, stream>>>(WIT_B, Wi_b, DD, H3);
  k_transpose_bf16<<<(HH * H3 + 255) / 256, 256, 0, stream>>>(WHT_F, Wh_f, HH, H3);
  k_transpose_bf16<<<(HH * H3 + 255) / 256, 256, 0, stream>>>(WHT_B, Wh_b, HH, H3);
  k_transpose_bf16<<<(H2 * HH + 255) / 256, 256, 0, stream>>>(WMVT, Wm, H2, HH);
  k_transpose_bf16<<<(H2 * HH + 255) / 256, 256, 0, stream>>>(WMVT + (size_t)HH * H2, Wv, H2, HH);

  // phase 1: input projections, both directions
  dim3 g1((BT / 16) * (H3 / 16) / 8, 2);
  k_gemm_in<<<g1, 256, 0, stream>>>(XBF, WIT_F, WIT_B, bi_f, bi_b, XG_F, XG_B);

  // phase 2: persistent recurrent scan (one WGP per direction)
  k_scan<<<2, 512, 0, stream>>>(XG_F, XG_B, WHT_F, WHT_B, bhnf, bhnb, OUTC);

  // phase 3: fused heads
  k_head<<<(BT / 16) * (H2 / 16) / 8, 256, 0, stream>>>(OUTC, WMVT, bm, bv, out);
}
